// RF_CRATE_11733850652900
// MI455X (gfx1250) — matrix operand order repaired and checked by reading, not compiled, not run on hardware
//
#include <hip/hip_runtime.h>

// ---------------------------------------------------------------------------
// Complex ViT forward for MI455X (gfx1250, wave32).
// All heavy GEMMs use v_wmma_f32_16x16x32_bf16 (bf16 in, f32 accum).
// Complex GEMM  = 4 real WMMA chains (3 accumulators) per 16x16 subtile.
// cgemm is register-blocked 32x32 per wave (2x2 subtiles): each k-step does
// 16 b128 loads -> 16 WMMAs (1.0 wmma/load). Epilogues fully unrolled to
// avoid m0/v_movrels dynamic-VGPR indexing.
// Activations: planar f32 (residual stream) + planar bf16 (WMMA inputs).
// ---------------------------------------------------------------------------

typedef __bf16 bf16_t;
typedef __bf16 v16bf __attribute__((ext_vector_type(16)));
typedef __bf16 v8bf  __attribute__((ext_vector_type(8)));
typedef float  v8f   __attribute__((ext_vector_type(8)));

#define B_      32
#define NTOK    197
#define NPAD    224            // token dim padded for K of attn 2nd GEMM
#define DIM_    384
#define HEADS_  6
#define DH_     64
#define DEPTH_  12
#define PDIM    768
#define NPATCH_ 196
#define MPATCH  (B_*NPATCH_)   // 6272
#define MTOK    (B_*NTOK)      // 6304
#define NCLS_   1000
#define NCLSP   1024           // padded to 32 for 32x32 tiling
#define MAT_    (DIM_*DIM_)
#define EPS_    1e-5f
#define SCALE_  0.125f
#define STEP_   0.1f
#define LAMBD_  0.1f
#define QTILES  13             // ceil(197/16)

// --------------------------- fragment loaders ------------------------------
// NT GEMM, both operands K-contiguous in memory.
// A 16x32 bf16 frag: lane<16 -> K 0..15 of row lane;
// lane>=16 -> K 16..31 of row lane-16 (matches B frag k order).
__device__ __forceinline__ v16bf load_frag_a(const bf16_t* tile, int ld, int lane) {
  const bf16_t* p = tile + (size_t)(lane & 15) * ld + ((lane >> 4) << 4);
  return *(const v16bf*)p;
}
// B 32x16 bf16 frag: lane<16 -> K 0..15 of column lane; lane>=16 -> K 16..31.
// With NT storage (N rows, K contiguous) this is 16 contiguous elements.
__device__ __forceinline__ v16bf load_frag_b(const bf16_t* tile, int ld, int lane) {
  const bf16_t* p = tile + (size_t)(lane & 15) * ld + ((lane >> 4) << 4);
  return *(const v16bf*)p;
}
__device__ __forceinline__ v8f wmma_bf16(v16bf a, v16bf b, v8f c) {
  return __builtin_amdgcn_wmma_f32_16x16x32_bf16(false, a, false, b, (short)0, c,
                                                 false, false);
}

// --------------------------- small helpers ---------------------------------
template<int NT>
__device__ __forceinline__ float4 blockSum4(float4 v, float4* buf) {
  int t = threadIdx.x;
  buf[t] = v;
  __syncthreads();
  for (int s = NT / 2; s > 0; s >>= 1) {
    if (t < s) {
      float4 o = buf[t + s]; float4 m = buf[t];
      m.x += o.x; m.y += o.y; m.z += o.z; m.w += o.w;
      buf[t] = m;
    }
    __syncthreads();
  }
  float4 r = buf[0];
  __syncthreads();
  return r;
}

// inv(csqrt(var_r + i*var_i + eps)) : complex LN denominator
__device__ __forceinline__ void cln_inv(float varr, float vari, float& invr, float& invi) {
  float a = varr + EPS_;
  float b = vari;
  float r = sqrtf(a * a + b * b);
  float t = sqrtf(0.5f * (r + a));
  float u = copysignf(sqrtf(fmaxf(0.5f * (r - a), 0.f)), b);
  float inv = 1.0f / fmaxf(r, 1e-30f);   // |sqrt(z)|^2 == r
  invr = t * inv;
  invi = -u * inv;
}

// --------------------------- conversion kernels ----------------------------
__global__ void k_cvt_bf16(const float* __restrict__ src, bf16_t* dR, bf16_t* dI,
                           int n, int npad) {
  int i = blockIdx.x * blockDim.x + threadIdx.x;
  if (i >= npad) return;
  float r = 0.f, im = 0.f;
  if (i < n) { r = src[2 * i]; im = src[2 * i + 1]; }
  dR[i] = (bf16_t)r; dI[i] = (bf16_t)im;
}
__global__ void k_cvt_f32(const float* __restrict__ src, float* dR, float* dI, int n) {
  int i = blockIdx.x * blockDim.x + threadIdx.x;
  if (i < n) { dR[i] = src[2 * i]; dI[i] = src[2 * i + 1]; }
}

// --------------------------- patchify + cln(ln1) ---------------------------
__global__ void __launch_bounds__(256) k_patchify(
    const float* __restrict__ imr, const float* __restrict__ imi,
    const float* __restrict__ w, const float* __restrict__ b,
    bf16_t* outR, bf16_t* outI) {
  int row = blockIdx.x;                  // 0..6271 = b*196 + patch
  int bb = row / NPATCH_;
  int pp = row % NPATCH_;
  int ph = pp / 14, pw = pp % 14;
  float xr[3], xi[3];
  float4 acc = make_float4(0, 0, 0, 0);
#pragma unroll
  for (int j = 0; j < 3; ++j) {
    int e = threadIdx.x + j * 256;       // e = py*48 + px*3 + c
    int py = e / 48, rem = e % 48, px = rem / 3, c = rem % 3;
    size_t idx = (((size_t)bb * 3 + c) * 224 + (ph * 16 + py)) * 224 + (pw * 16 + px);
    float r = imr[idx], im = imi[idx];
    xr[j] = r; xi[j] = im;
    acc.x += r; acc.y += im; acc.z += r * r; acc.w += im * im;
  }
  __shared__ float4 buf[256];
  float4 s = blockSum4<256>(acc, buf);
  const float invL = 1.f / 768.f;
  float mr = s.x * invL, mi = s.y * invL;
  float vr = s.z * invL - mr * mr, vi = s.w * invL - mi * mi;
  float ir, ii; cln_inv(vr, vi, ir, ii);
#pragma unroll
  for (int j = 0; j < 3; ++j) {
    int e = threadIdx.x + j * 256;
    float dr = xr[j] - mr, di = xi[j] - mi;
    float nr = dr * ir - di * ii, ni = dr * ii + di * ir;
    float wr = w[2 * e], wi = w[2 * e + 1], br = b[2 * e], bi = b[2 * e + 1];
    float ov = wr * nr - wi * ni + br;
    float oi = wr * ni + wi * nr + bi;
    outR[(size_t)row * PDIM + e] = (bf16_t)ov;
    outI[(size_t)row * PDIM + e] = (bf16_t)oi;
  }
}

// ------------- patch_ln2 + cls concat + pos_emb -> residual stream x -------
__global__ void __launch_bounds__(128) k_assemble(
    const float* __restrict__ Yr, const float* __restrict__ Yi,
    const float* __restrict__ w, const float* __restrict__ b,
    const float* __restrict__ clsp, const float* __restrict__ pos,
    float* xr, float* xi) {
  int tok = blockIdx.x;                 // 0..6303
  int bb = tok / NTOK, n = tok % NTOK;
  int t = threadIdx.x;
  __shared__ float4 buf[128];
  if (n == 0) {                         // cls token (no LN), +pos[0]
#pragma unroll
    for (int j = 0; j < 3; ++j) {
      int e = t + j * 128;
      xr[(size_t)tok * DIM_ + e] = clsp[2 * e]     + pos[2 * e];
      xi[(size_t)tok * DIM_ + e] = clsp[2 * e + 1] + pos[2 * e + 1];
    }
    return;                             // block-uniform exit
  }
  size_t row = (size_t)bb * NPATCH_ + (n - 1);
  float lr[3], li[3];
  float4 acc = make_float4(0, 0, 0, 0);
#pragma unroll
  for (int j = 0; j < 3; ++j) {
    int e = t + j * 128;
    float r = Yr[row * DIM_ + e], im = Yi[row * DIM_ + e];
    lr[j] = r; li[j] = im;
    acc.x += r; acc.y += im; acc.z += r * r; acc.w += im * im;
  }
  float4 s = blockSum4<128>(acc, buf);
  const float invL = 1.f / 384.f;
  float mr = s.x * invL, mi = s.y * invL;
  float vr = s.z * invL - mr * mr, vi = s.w * invL - mi * mi;
  float ir, ii; cln_inv(vr, vi, ir, ii);
#pragma unroll
  for (int j = 0; j < 3; ++j) {
    int e = t + j * 128;
    float dr = lr[j] - mr, di = li[j] - mi;
    float nr = dr * ir - di * ii, ni = dr * ii + di * ir;
    float wr = w[2 * e], wi = w[2 * e + 1], br = b[2 * e], bi = b[2 * e + 1];
    size_t pe = (size_t)n * DIM_ + e;
    xr[(size_t)tok * DIM_ + e] = wr * nr - wi * ni + br + pos[2 * pe];
    xi[(size_t)tok * DIM_ + e] = wr * ni + wi * nr + bi + pos[2 * pe + 1];
  }
}

// --------------------------- generic complex LN ----------------------------
__global__ void __launch_bounds__(128) k_cln(
    const float* __restrict__ xr, const float* __restrict__ xi,
    const float* __restrict__ w, const float* __restrict__ b,
    bf16_t* oR, bf16_t* oI, float* oFr, float* oFi, int rowMul) {
  int orow = blockIdx.x;
  size_t srow = (size_t)orow * rowMul;
  const float* pr = xr + srow * DIM_;
  const float* pi = xi + srow * DIM_;
  int t = threadIdx.x;
  float lr[3], li[3];
  float4 acc = make_float4(0, 0, 0, 0);
#pragma unroll
  for (int j = 0; j < 3; ++j) {
    int e = t + j * 128;
    float r = pr[e], im = pi[e];
    lr[j] = r; li[j] = im;
    acc.x += r; acc.y += im; acc.z += r * r; acc.w += im * im;
  }
  __shared__ float4 buf[128];
  float4 s = blockSum4<128>(acc, buf);
  const float invL = 1.f / 384.f;
  float mr = s.x * invL, mi = s.y * invL;
  float vr = s.z * invL - mr * mr, vi = s.w * invL - mi * mi;
  float ir, ii; cln_inv(vr, vi, ir, ii);
#pragma unroll
  for (int j = 0; j < 3; ++j) {
    int e = t + j * 128;
    float dr = lr[j] - mr, di = li[j] - mi;
    float nr = dr * ir - di * ii, ni = dr * ii + di * ir;
    float wr = w[2 * e], wi = w[2 * e + 1], br = b[2 * e], bi = b[2 * e + 1];
    float ov = wr * nr - wi * ni + br;
    float oi = wr * ni + wi * nr + bi;
    size_t o = (size_t)orow * DIM_ + e;
    oR[o] = (bf16_t)ov; oI[o] = (bf16_t)oi;
    if (oFr) { oFr[o] = ov; oFi[o] = oi; }   // f32 copy (safe in-place: own elems)
  }
}

// --------------------------- generic complex WMMA GEMM ---------------------
// Y = A @ B^T (complex, NT form), per-wave 32x32 tile (2x2 16x16 subtiles).
// Epilogue modes:
//  0: f32 out + bias            1: qkv scatter -> w(b,h,224,64) + wT(b,h,64,224)
//  2: ff:  crelu(res + 0.1*v - 0.01)  (res = normed gx)
//  3: out-proj: v + bias + res  4: head: |v + bias| -> d_out (col<1000)
__device__ __forceinline__ void cgemm_epilogue16(
    v8f accRR, v8f accII, v8f accI, int m0, int n0, int lane, int ldc,
    float* Cr, float* Ci,
    const float* __restrict__ biasR, const float* __restrict__ biasI,
    const float* __restrict__ resR, const float* __restrict__ resI,
    bf16_t* O1r, bf16_t* O1i, bf16_t* O2r, bf16_t* O2i,
    float* outAbs, int mode) {
  int col = n0 + (lane & 15);
  int rbase = m0 + ((lane >> 4) << 3);
#pragma unroll
  for (int r = 0; r < 8; ++r) {          // constant r: no v_movrels
    int row = rbase + r;
    float vr = accRR[r] - accII[r];
    float vi = accI[r];
    size_t o = (size_t)row * ldc + col;
    if (mode == 0) {
      Cr[o] = vr + biasR[col];
      Ci[o] = vi + biasI[col];
    } else if (mode == 1) {
      int bb = row / NTOK, n = row % NTOK;
      int h = col >> 6, dh = col & 63;
      size_t bh = (size_t)bb * HEADS_ + h;
      O1r[(bh * NPAD + n) * DH_ + dh] = (bf16_t)vr;
      O1i[(bh * NPAD + n) * DH_ + dh] = (bf16_t)vi;
      O2r[(bh * DH_ + dh) * NPAD + n] = (bf16_t)vr;
      O2i[(bh * DH_ + dh) * NPAD + n] = (bf16_t)vi;
    } else if (mode == 2) {
      float ov = resR[o] + STEP_ * vr - STEP_ * LAMBD_;   // -0.01 real only
      float oi = resI[o] + STEP_ * vi;
      Cr[o] = fmaxf(ov, 0.f);
      Ci[o] = fmaxf(oi, 0.f);
    } else if (mode == 3) {
      Cr[o] = vr + biasR[col] + resR[o];
      Ci[o] = vi + biasI[col] + resI[o];
    } else {   // mode 4: head -> |logits|
      if (col < NCLS_) {
        float rr = vr + biasR[col], ii = vi + biasI[col];
        outAbs[(size_t)row * NCLS_ + col] = sqrtf(rr * rr + ii * ii);
      }
    }
  }
}

__global__ void __launch_bounds__(128) k_cgemm(
    const bf16_t* __restrict__ Ar, const bf16_t* __restrict__ Ai,
    const bf16_t* __restrict__ Br, const bf16_t* __restrict__ Bi,
    int Mtiles, int Ntiles, int K, int lda, int ldb, int ldc,   // 32x32 tiles
    float* Cr, float* Ci,
    const float* __restrict__ biasR, const float* __restrict__ biasI,
    const float* __restrict__ resR, const float* __restrict__ resI,
    bf16_t* O1r, bf16_t* O1i, bf16_t* O2r, bf16_t* O2i,
    float* outAbs, int mode) {
  int lane = threadIdx.x & 31;
  int wave = threadIdx.x >> 5;
  int tile = blockIdx.x * 4 + wave;
  if (tile >= Mtiles * Ntiles) return;         // wave-uniform
  int mt = tile / Ntiles, nt = tile % Ntiles;
  int m0 = mt * 32, n0 = nt * 32;
  v8f accRR[2][2] = {}; v8f accII[2][2] = {}; v8f accI[2][2] = {};
  for (int k0 = 0; k0 < K; k0 += 32) {
    v16bf ar[2], ai[2], br[2], bi[2];
#pragma unroll
    for (int i = 0; i < 2; ++i) {
      ar[i] = load_frag_a(Ar + (size_t)(m0 + 16 * i) * lda + k0, lda, lane);
      ai[i] = load_frag_a(Ai + (size_t)(m0 + 16 * i) * lda + k0, lda, lane);
      br[i] = load_frag_b(Br + (size_t)(n0 + 16 * i) * ldb + k0, ldb, lane);
      bi[i] = load_frag_b(Bi + (size_t)(n0 + 16 * i) * ldb + k0, ldb, lane);
    }
#pragma unroll
    for (int i = 0; i < 2; ++i)
#pragma unroll
      for (int j = 0; j < 2; ++j) {
        accRR[i][j] = wmma_bf16(ar[i], br[j], accRR[i][j]);   // Re += ar*br
        accII[i][j] = wmma_bf16(ai[i], bi[j], accII[i][j]);   // Re -= ai*bi
        accI [i][j] = wmma_bf16(ar[i], bi[j], accI [i][j]);   // Im += ar*bi
        accI [i][j] = wmma_bf16(ai[i], br[j], accI [i][j]);   // Im += ai*br
      }
  }
#pragma unroll
  for (int i = 0; i < 2; ++i)
#pragma unroll
    for (int j = 0; j < 2; ++j)
      cgemm_epilogue16(accRR[i][j], accII[i][j], accI[i][j],
                       m0 + 16 * i, n0 + 16 * j, lane, ldc,
                       Cr, Ci, biasR, biasI, resR, resI,
                       O1r, O1i, O2r, O2i, outAbs, mode);
}

// --------------------------- fused complex attention -----------------------
// One block (4 waves) per (b, h, 16-query tile).
// Phase1: S = q . conj(w)^T via WMMA -> LDS (f32).
// Phase2: complex softmax over |S| (valid keys only), A -> LDS bf16 (pad=0).
// Phase3: O = A @ w via WMMA (K = 224, wT layout), masked store -> attn bf16.
__global__ void __launch_bounds__(128) k_attn(
    const bf16_t* __restrict__ Wr, const bf16_t* __restrict__ Wi,
    const bf16_t* __restrict__ WTr, const bf16_t* __restrict__ WTi,
    bf16_t* Or, bf16_t* Oi) {
  __shared__ float Sr[16][NPAD];
  __shared__ float Si[16][NPAD];
  __shared__ __align__(32) bf16_t Par[16][NPAD];
  __shared__ __align__(32) bf16_t Pai[16][NPAD];
  int bh = blockIdx.x / QTILES;
  int qt = blockIdx.x % QTILES;
  int bb = bh / HEADS_, h = bh % HEADS_;
  int lane = threadIdx.x & 31, wave = threadIdx.x >> 5;
  int n0 = qt * 16;
  const bf16_t* wr = Wr + (size_t)bh * NPAD * DH_;
  const bf16_t* wi = Wi + (size_t)bh * NPAD * DH_;
  // ---- phase 1: scores ----
  v16bf qr0 = load_frag_a(wr + (size_t)n0 * DH_,      DH_, lane);
  v16bf qr1 = load_frag_a(wr + (size_t)n0 * DH_ + 32, DH_, lane);
  v16bf qi0 = load_frag_a(wi + (size_t)n0 * DH_,      DH_, lane);
  v16bf qi1 = load_frag_a(wi + (size_t)n0 * DH_ + 32, DH_, lane);
  for (int mt = wave; mt < NPAD / 16; mt += 4) {
    int m0 = mt * 16;
    v16bf kr0 = load_frag_b(wr + (size_t)m0 * DH_,      DH_, lane);
    v16bf kr1 = load_frag_b(wr + (size_t)m0 * DH_ + 32, DH_, lane);
    v16bf ki0 = load_frag_b(wi + (size_t)m0 * DH_,      DH_, lane);
    v16bf ki1 = load_frag_b(wi + (size_t)m0 * DH_ + 32, DH_, lane);
    // S = q*conj(k): Re = qr*kr + qi*ki ; Im = qi*kr - qr*ki
    v8f aR = {}; v8f aI1 = {}; v8f aI2 = {};
    aR  = wmma_bf16(qr0, kr0, aR);  aR  = wmma_bf16(qr1, kr1, aR);
    aR  = wmma_bf16(qi0, ki0, aR);  aR  = wmma_bf16(qi1, ki1, aR);
    aI1 = wmma_bf16(qi0, kr0, aI1); aI1 = wmma_bf16(qi1, kr1, aI1);
    aI2 = wmma_bf16(qr0, ki0, aI2); aI2 = wmma_bf16(qr1, ki1, aI2);
    int colr = m0 + (lane & 15);
    int rb = (lane >> 4) << 3;
#pragma unroll
    for (int r = 0; r < 8; ++r) {
      Sr[rb + r][colr] = SCALE_ * aR[r];
      Si[rb + r][colr] = SCALE_ * (aI1[r] - aI2[r]);
    }
  }
  __syncthreads();
  // ---- phase 2: complex softmax, 8 threads per query row ----
  {
    int row = threadIdx.x >> 3, sub = threadIdx.x & 7;
    float mx = -1e30f;
    for (int m = sub; m < NTOK; m += 8) {
      float sr = Sr[row][m], si = Si[row][m];
      mx = fmaxf(mx, sqrtf(sr * sr + si * si));
    }
    mx = fmaxf(mx, __shfl_xor(mx, 1));
    mx = fmaxf(mx, __shfl_xor(mx, 2));
    mx = fmaxf(mx, __shfl_xor(mx, 4));
    float sum = 0.f;
    for (int m = sub; m < NTOK; m += 8) {
      float sr = Sr[row][m], si = Si[row][m];
      sum += __expf(sqrtf(sr * sr + si * si) - mx);
    }
    sum += __shfl_xor(sum, 1);
    sum += __shfl_xor(sum, 2);
    sum += __shfl_xor(sum, 4);
    float inv = 1.f / sum;
    for (int m = sub; m < NTOK; m += 8) {
      float sr = Sr[row][m], si = Si[row][m];
      float mg = sqrtf(sr * sr + si * si);
      float pw = __expf(mg - mx) * inv;
      float ar, ai;
      if (mg > 0.f) { float sc = pw / mg; ar = sr * sc; ai = si * sc; }
      else          { ar = pw; ai = 0.f; }       // angle(0)=0 -> exp(i0)=1
      Par[row][m] = (bf16_t)ar;
      Pai[row][m] = (bf16_t)ai;
    }
    for (int m = NTOK + sub; m < NPAD; m += 8) { // zero K-padding
      Par[row][m] = (bf16_t)0.f;
      Pai[row][m] = (bf16_t)0.f;
    }
  }
  __syncthreads();
  // ---- phase 3: O = A @ w, wave handles one 16-wide dhead tile ----
  {
    int d0 = wave * 16;
    const bf16_t* wtr = WTr + ((size_t)bh * DH_ + d0) * NPAD;
    const bf16_t* wti = WTi + ((size_t)bh * DH_ + d0) * NPAD;
    v8f oRR = {}; v8f oII = {}; v8f oI = {};
#pragma unroll
    for (int k0 = 0; k0 < NPAD; k0 += 32) {
      v16bf ar = load_frag_a(&Par[0][0] + k0, NPAD, lane);
      v16bf ai = load_frag_a(&Pai[0][0] + k0, NPAD, lane);
      v16bf br = load_frag_b(wtr + k0, NPAD, lane);
      v16bf bi = load_frag_b(wti + k0, NPAD, lane);
      oRR = wmma_bf16(ar, br, oRR);
      oII = wmma_bf16(ai, bi, oII);
      oI  = wmma_bf16(ar, bi, oI);
      oI  = wmma_bf16(ai, br, oI);
    }
    int dh = d0 + (lane & 15);
    int rb = (lane >> 4) << 3;
#pragma unroll
    for (int r = 0; r < 8; ++r) {
      int n = n0 + rb + r;
      if (n < NTOK) {
        size_t o = ((size_t)bb * NTOK + n) * DIM_ + h * DH_ + dh;
        Or[o] = (bf16_t)(oRR[r] - oII[r]);
        Oi[o] = (bf16_t)oI[r];
      }
    }
  }
}

// ===========================================================================
extern "C" void kernel_launch(void* const* d_in, const int* in_sizes, int n_in,
                              void* d_out, int out_size, void* d_ws, size_t ws_size,
                              hipStream_t stream) {
  (void)in_sizes; (void)n_in; (void)out_size; (void)ws_size;
  // ---- d_in layout: setup_inputs() insertion order, complex64 interleaved ----
  const float* img_r  = (const float*)d_in[0];
  const float* img_i  = (const float*)d_in[1];
  const float* p_ln1w = (const float*)d_in[2];
  const float* p_ln1b = (const float*)d_in[3];
  const float* p_linw = (const float*)d_in[4];
  const float* p_linb = (const float*)d_in[5];
  const float* p_ln2w = (const float*)d_in[6];
  const float* p_ln2b = (const float*)d_in[7];
  const float* p_pos  = (const float*)d_in[8];
  const float* p_cls  = (const float*)d_in[9];
  // layers: 10 + 8*L : ln1_w, ln1_b, qkv_w, out_w, out_b, ln2_w, ln2_b, ff_w
  const float* head_lnw = (const float*)d_in[10 + 8 * DEPTH_ + 0];
  const float* head_lnb = (const float*)d_in[10 + 8 * DEPTH_ + 1];
  const float* head_w   = (const float*)d_in[10 + 8 * DEPTH_ + 2];
  const float* head_b   = (const float*)d_in[10 + 8 * DEPTH_ + 3];

  // ---- workspace allocator ----
  char* wp = (char*)d_ws;
  auto alloc = [&](size_t bytes) -> char* {
    char* r = wp; wp += (bytes + 255) & ~(size_t)255; return r;
  };
  bf16_t* wpR = (bf16_t*)alloc((size_t)DIM_ * PDIM * 2);
  bf16_t* wpI = (bf16_t*)alloc((size_t)DIM_ * PDIM * 2);
  bf16_t* wqR = (bf16_t*)alloc((size_t)DEPTH_ * MAT_ * 2);
  bf16_t* wqI = (bf16_t*)alloc((size_t)DEPTH_ * MAT_ * 2);
  bf16_t* woR = (bf16_t*)alloc((size_t)DEPTH_ * MAT_ * 2);
  bf16_t* woI = (bf16_t*)alloc((size_t)DEPTH_ * MAT_ * 2);
  bf16_t* wfR = (bf16_t*)alloc((size_t)DEPTH_ * MAT_ * 2);
  bf16_t* wfI = (bf16_t*)alloc((size_t)DEPTH_ * MAT_ * 2);
  bf16_t* whR = (bf16_t*)alloc((size_t)NCLSP * DIM_ * 2);
  bf16_t* whI = (bf16_t*)alloc((size_t)NCLSP * DIM_ * 2);
  float* bpR = (float*)alloc(DIM_ * 4);  float* bpI = (float*)alloc(DIM_ * 4);
  float* boR = (float*)alloc((size_t)DEPTH_ * DIM_ * 4);
  float* boI = (float*)alloc((size_t)DEPTH_ * DIM_ * 4);
  float* bhR = (float*)alloc(NCLS_ * 4); float* bhI = (float*)alloc(NCLS_ * 4);
  bf16_t* XpR = (bf16_t*)alloc((size_t)MPATCH * PDIM * 2);
  bf16_t* XpI = (bf16_t*)alloc((size_t)MPATCH * PDIM * 2);
  float* YpR = (float*)alloc((size_t)MPATCH * DIM_ * 4);
  float* YpI = (float*)alloc((size_t)MPATCH * DIM_ * 4);
  float* xR  = (float*)alloc((size_t)MTOK * DIM_ * 4);
  float* xI  = (float*)alloc((size_t)MTOK * DIM_ * 4);
  bf16_t* xnR = (bf16_t*)alloc((size_t)MTOK * DIM_ * 2);
  bf16_t* xnI = (bf16_t*)alloc((size_t)MTOK * DIM_ * 2);
  size_t wsz = (size_t)B_ * HEADS_ * NPAD * DH_;
  bf16_t* wR  = (bf16_t*)alloc(wsz * 2);  bf16_t* wI  = (bf16_t*)alloc(wsz * 2);
  bf16_t* wTR = (bf16_t*)alloc(wsz * 2);  bf16_t* wTI = (bf16_t*)alloc(wsz * 2);
  bf16_t* atR = (bf16_t*)alloc((size_t)MTOK * DIM_ * 2);
  bf16_t* atI = (bf16_t*)alloc((size_t)MTOK * DIM_ * 2);
  float* gxR = (float*)alloc((size_t)MTOK * DIM_ * 4);
  float* gxI = (float*)alloc((size_t)MTOK * DIM_ * 4);
  bf16_t* xhR = (bf16_t*)alloc((size_t)B_ * DIM_ * 2);
  bf16_t* xhI = (bf16_t*)alloc((size_t)B_ * DIM_ * 2);

  auto cvtb = [&](const float* src, bf16_t* dR, bf16_t* dI, int n, int npad) {
    k_cvt_bf16<<<(npad + 255) / 256, 256, 0, stream>>>(src, dR, dI, n, npad);
  };
  auto cvtf = [&](const float* src, float* dR, float* dI, int n) {
    k_cvt_f32<<<(n + 255) / 256, 256, 0, stream>>>(src, dR, dI, n);
  };
  // ---- weight/bias conversion (interleaved c64 -> planar bf16/f32) ----
  cvtb(p_linw, wpR, wpI, DIM_ * PDIM, DIM_ * PDIM);
  cvtf(p_linb, bpR, bpI, DIM_);
  for (int L = 0; L < DEPTH_; ++L) {
    const float* qkvw = (const float*)d_in[10 + 8 * L + 2];
    const float* outw = (const float*)d_in[10 + 8 * L + 3];
    const float* outb = (const float*)d_in[10 + 8 * L + 4];
    const float* ffw  = (const float*)d_in[10 + 8 * L + 7];
    cvtb(qkvw, wqR + (size_t)L * MAT_, wqI + (size_t)L * MAT_, MAT_, MAT_);
    cvtb(outw, woR + (size_t)L * MAT_, woI + (size_t)L * MAT_, MAT_, MAT_);
    cvtb(ffw,  wfR + (size_t)L * MAT_, wfI + (size_t)L * MAT_, MAT_, MAT_);
    cvtf(outb, boR + (size_t)L * DIM_, boI + (size_t)L * DIM_, DIM_);
  }
  cvtb(head_w, whR, whI, NCLS_ * DIM_, NCLSP * DIM_);   // pad rows 1000..1023 = 0
  cvtf(head_b, bhR, bhI, NCLS_);

  auto gemm = [&](const bf16_t* Ar, const bf16_t* Ai,
                  const bf16_t* Br, const bf16_t* Bi,
                  int Mt, int Nt, int K, int lda, int ldb, int ldc,
                  float* Cr, float* Ci, const float* bR, const float* bI,
                  const float* rR, const float* rI,
                  bf16_t* o1r, bf16_t* o1i, bf16_t* o2r, bf16_t* o2i,
                  float* oabs, int mode) {
    int total = Mt * Nt;     // Mt/Nt are counts of 32x32 tiles
    k_cgemm<<<(total + 3) / 4, 128, 0, stream>>>(
        Ar, Ai, Br, Bi, Mt, Nt, K, lda, ldb, ldc, Cr, Ci,
        bR, bI, rR, rI, o1r, o1i, o2r, o2i, oabs, mode);
  };

  // ---- patch embed ----
  k_patchify<<<MPATCH, 256, 0, stream>>>(img_r, img_i, p_ln1w, p_ln1b, XpR, XpI);
  gemm(XpR, XpI, wpR, wpI, MPATCH / 32, DIM_ / 32, PDIM, PDIM, PDIM, DIM_,
       YpR, YpI, bpR, bpI, nullptr, nullptr,
       nullptr, nullptr, nullptr, nullptr, nullptr, 0);
  k_assemble<<<MTOK, 128, 0, stream>>>(YpR, YpI, p_ln2w, p_ln2b, p_cls, p_pos, xR, xI);

  // ---- transformer layers ----
  for (int L = 0; L < DEPTH_; ++L) {
    const float* ln1w = (const float*)d_in[10 + 8 * L + 0];
    const float* ln1b = (const float*)d_in[10 + 8 * L + 1];
    const float* ln2w = (const float*)d_in[10 + 8 * L + 5];
    const float* ln2b = (const float*)d_in[10 + 8 * L + 6];
    // ln1 -> bf16
    k_cln<<<MTOK, 128, 0, stream>>>(xR, xI, ln1w, ln1b, xnR, xnI,
                                    nullptr, nullptr, 1);
    // qkv GEMM -> per-head w & wT (bf16)
    gemm(xnR, xnI, wqR + (size_t)L * MAT_, wqI + (size_t)L * MAT_,
         MTOK / 32, DIM_ / 32, DIM_, DIM_, DIM_, DIM_,
         nullptr, nullptr, nullptr, nullptr, nullptr, nullptr,
         wR, wI, wTR, wTI, nullptr, 1);
    // fused attention
    k_attn<<<B_ * HEADS_ * QTILES, 128, 0, stream>>>(wR, wI, wTR, wTI, atR, atI);
    // out-proj + bias + residual(x) -> gx
    gemm(atR, atI, woR + (size_t)L * MAT_, woI + (size_t)L * MAT_,
         MTOK / 32, DIM_ / 32, DIM_, DIM_, DIM_, DIM_,
         gxR, gxI, boR + (size_t)L * DIM_, boI + (size_t)L * DIM_,
         xR, xI, nullptr, nullptr, nullptr, nullptr, nullptr, 3);
    // ln2 -> bf16 + f32 (f32 copy in-place over gx; needed by ff residual)
    k_cln<<<MTOK, 128, 0, stream>>>(gxR, gxI, ln2w, ln2b, xnR, xnI,
                                    gxR, gxI, 1);
    // ff + crelu(xn + 0.1*x1 - 0.01) -> new x
    gemm(xnR, xnI, wfR + (size_t)L * MAT_, wfI + (size_t)L * MAT_,
         MTOK / 32, DIM_ / 32, DIM_, DIM_, DIM_, DIM_,
         xR, xI, nullptr, nullptr, gxR, gxI,
         nullptr, nullptr, nullptr, nullptr, nullptr, 2);
  }

  // ---- head: gather token 0, LN, GEMM -> |logits| ----
  k_cln<<<B_, 128, 0, stream>>>(xR, xI, head_lnw, head_lnb, xhR, xhI,
                                nullptr, nullptr, NTOK);
  gemm(xhR, xhI, whR, whI, B_ / 32, NCLSP / 32, DIM_, DIM_, DIM_, NCLS_,
       nullptr, nullptr, bhR, bhI, nullptr, nullptr,
       nullptr, nullptr, nullptr, nullptr, (float*)d_out, 4);
}
